// NodeTaskHead_35107062677959
// MI455X (gfx1250) — matrix operand order repaired and checked by reading, not compiled, not run on hardware
//
#include <hip/hip_runtime.h>

typedef __attribute__((ext_vector_type(16))) __bf16 v16bf;
typedef __attribute__((ext_vector_type(8)))  float  v8f;

constexpr int Bb = 4, Nn = 512, Ee = 256, Hh = 8, Dd = 32;

#if defined(__HIP_DEVICE_COMPILE__) && defined(__gfx1250__)
#define HAVE_GFX1250 1
#else
#define HAVE_GFX1250 0
#endif

#if defined(__HIP_DEVICE_COMPILE__) && \
    __has_builtin(__builtin_amdgcn_cvt_pk_bf16_f32)
#define HAVE_PK 1
#else
#define HAVE_PK 0
#endif

__device__ __forceinline__ __bf16 f2bf(float f) {
    unsigned u = __builtin_bit_cast(unsigned, f);
    u += 0x7FFFu + ((u >> 16) & 1u);            // round-to-nearest-even
    unsigned short h = (unsigned short)(u >> 16);
    return __builtin_bit_cast(__bf16, h);
}

// pack two f32 -> two bf16 in one VALU op (v_cvt_pk_bf16_f32, RNE)
__device__ __forceinline__ void pack2(float a, float b, __bf16* dst) {
#if HAVE_PK
    auto t = __builtin_amdgcn_cvt_pk_bf16_f32(a, b);
    __builtin_memcpy(dst, &t, 4);
#elif HAVE_GFX1250
    unsigned r;
    asm("v_cvt_pk_bf16_f32 %0, %1, %2" : "=v"(r) : "v"(a), "v"(b));
    __builtin_memcpy(dst, &r, 4);
#else
    dst[0] = f2bf(a); dst[1] = f2bf(b);
#endif
}

__device__ __forceinline__ void cvt8(const float* __restrict__ p, __bf16* dst) {
    float4 a = *(const float4*)p;
    float4 b = *(const float4*)(p + 4);
    pack2(a.x, a.y, dst);     pack2(a.z, a.w, dst + 2);
    pack2(b.x, b.y, dst + 4); pack2(b.z, b.w, dst + 6);
}

// 16-byte global -> LDS copy: gfx1250 async-to-LDS (ASYNCcnt), inline asm.
// VDST = LDS byte address (low 32 bits of generic pointer), VADDR = 64b global.
__device__ __forceinline__ void cp16(const void* g, void* l) {
#if HAVE_GFX1250
    unsigned ldsoff = (unsigned)(size_t)l;
    asm volatile("global_load_async_to_lds_b128 %0, %1, off"
                 :: "v"(ldsoff), "v"(g) : "memory");
#else
    float4 t = *(const float4*)g;
    *(float4*)l = t;
#endif
}

template <int N>
__device__ __forceinline__ void wait_async() {
#if HAVE_GFX1250
#if __has_builtin(__builtin_amdgcn_s_wait_asynccnt)
    __builtin_amdgcn_s_wait_asynccnt(N);
#else
    asm volatile("s_wait_asynccnt %0" :: "i"(N) : "memory");
#endif
#endif
}

// ---------------------------------------------------------------------------
// Kernel 1: Q/K/V projection GEMM [2048x256]@[256x256]^T, bf16 WMMA, f32 acc.
// q scaled by D^-0.5; v stored transposed [b,h,d,n].
// ---------------------------------------------------------------------------
__global__ __launch_bounds__(32)
void qkv_kernel(const float* __restrict__ qin,
                const float* __restrict__ Wq, const float* __restrict__ bq,
                const float* __restrict__ Wk, const float* __restrict__ bk,
                const float* __restrict__ Wv, const float* __restrict__ bv,
                __bf16* __restrict__ q_s, __bf16* __restrict__ k_m,
                __bf16* __restrict__ v_t) {
    const int lane = threadIdx.x, lh = lane >> 4, ll = lane & 15;
    const int rt = blockIdx.x, ct = blockIdx.y, w = blockIdx.z;
    const float* W    = (w == 0) ? Wq : (w == 1) ? Wk : Wv;
    const float* bias = (w == 0) ? bq : (w == 1) ? bk : bv;

    const int arow = rt * 16 + ll;
    const int bcol = ct * 16 + ll;
    v8f acc = {};
    for (int kk = 0; kk < 8; ++kk) {
        __bf16 abuf[16], bbuf[16];
        const float* ap = qin + (size_t)arow * Ee + kk * 32 + lh * 8;
        cvt8(ap,      abuf);               // K = kk*32 + 8*lh + [0..7]
        cvt8(ap + 16, abuf + 8);           // K = +16
        const float* bp = W + (size_t)bcol * Ee + kk * 32 + lh * 8;
        cvt8(bp,     bbuf);                // K = kk*32 + 8*lh + [0..7]
        cvt8(bp + 16, bbuf + 8);
        v16bf A, Bf;
        __builtin_memcpy(&A,  abuf, 32);
        __builtin_memcpy(&Bf, bbuf, 32);
        acc = __builtin_amdgcn_wmma_f32_16x16x32_bf16(false, A, false, Bf,
                                                      (short)0, acc, false, false);
    }
    const float bcv = bias[bcol];
    const float scale = 0.17677669529663687f;   // 1/sqrt(D)
    const int h = bcol >> 5, d = bcol & 31;
#pragma unroll
    for (int i = 0; i < 8; ++i) {
        int r = rt * 16 + i + 8 * lh;           // global row = b*N + n
        int b = r >> 9, n = r & 511;
        float val = acc[i] + bcv;
        if (w == 0)      q_s[(((size_t)(b * Hh + h) * Nn + n) * Dd) + d] = f2bf(val * scale);
        else if (w == 1) k_m[(((size_t)(b * Hh + h) * Nn + n) * Dd) + d] = f2bf(val);
        else             v_t[(((size_t)(b * Hh + h) * Dd + d) * Nn) + n] = f2bf(val);
    }
}

// ---------------------------------------------------------------------------
// Kernel 2: initialize output with the per-channel bias
// ---------------------------------------------------------------------------
__global__ void init_out_kernel(float* __restrict__ out,
                                const float* __restrict__ bf1,
                                const float* __restrict__ bf2,
                                const float* __restrict__ bf3) {
    int i = blockIdx.x * blockDim.x + threadIdx.x;
    if (i < Bb * Nn * 3) {
        int c = i % 3;
        out[i] = (c == 0) ? *bf1 : (c == 1) ? *bf2 : *bf3;
    }
}

// ---------------------------------------------------------------------------
// Kernel 3: fused logits -> softmax(+mask fold) -> (probs*delta)@v -> dot(Wf)
// One block per (b*h, 16-row n-tile); 3 waves, wave index = channel c.
// delta_pos tiles double-buffered through LDS via async global->LDS b128.
// ---------------------------------------------------------------------------
__global__ __launch_bounds__(96)
void attn_fused_kernel(const __bf16* __restrict__ q_s, const __bf16* __restrict__ k_m,
                       const __bf16* __restrict__ v_t,
                       const float* __restrict__ attn_bias,
                       const float* __restrict__ delta_pos,
                       const unsigned char* __restrict__ dmask,
                       const int* __restrict__ doa_p,
                       const float* __restrict__ Wf1, const float* __restrict__ Wf2,
                       const float* __restrict__ Wf3, float* __restrict__ out) {
    constexpr int LDSTR = 516;                     // probs row pad (bank spread)
    constexpr int DROW  = 100;                     // delta row pad: 96 -> 100
    constexpr int MROW  = 528;                     // mask row pad (16B aligned)
    __shared__ float probs[16 * LDSTR];
    __shared__ float dbuf[2][16 * DROW];
    __shared__ unsigned char mbuf[16 * MROW];
    __shared__ float red[16 * 6];
    __shared__ float rowstat[16];

    const int tid = threadIdx.x;
    const int wave = tid >> 5;                     // 0..2 == channel c
    const int lane = tid & 31, lh = lane >> 4, ll = lane & 15;
    const int bh = blockIdx.y;                     // b*H + h
    const int b = bh >> 3, h = bh & 7;
    const int n0 = blockIdx.x * 16;
    const int doa = *doa_p;

    // ---- prologue: kick mask tile + delta chunk 0 into LDS (async) -----------
    const float* dpb = delta_pos + (size_t)(b * Nn + n0) * Nn * 3;
    const unsigned char* mkb = dmask + (size_t)n0 * Nn;
    {
#pragma unroll
        for (int r = 0; r < 6; ++r) {              // 512 16B units of mask
            int u = tid + r * 96;
            if (u < 512) {
                int nl = u >> 5, o = u & 31;
                cp16(mkb + (size_t)nl * Nn + o * 16, &mbuf[nl * MROW + o * 16]);
            }
        }
        const int nl0 = tid / 24, o16 = tid % 24;  // 96 threads, 384 16B units
#pragma unroll
        for (int r = 0; r < 4; ++r) {
            int nl = nl0 + 4 * r;
            cp16(dpb + (size_t)nl * (Nn * 3) + o16 * 4, &dbuf[0][nl * DROW + o16 * 4]);
        }
    }

    // ---- stage 1: logits = q @ k^T + bias -> LDS (f32) -----------------------
    v16bf Aq;
    {
        const __bf16* qb = q_s + ((size_t)bh * Nn + (n0 + ll)) * Dd;
        __bf16 abuf[16];
        uint4 u0 = *(const uint4*)(qb + lh * 8);
        uint4 u1 = *(const uint4*)(qb + 16 + lh * 8);
        __builtin_memcpy(abuf,     &u0, 16);
        __builtin_memcpy(abuf + 8, &u1, 16);
        __builtin_memcpy(&Aq, abuf, 32);
    }
    const float* biasrow = attn_bias + (size_t)bh * Nn * Nn;
    for (int mt = wave; mt < 32; mt += 3) {
        const __bf16* kb = k_m + ((size_t)bh * Nn + (mt * 16 + ll)) * Dd + lh * 8;
        __bf16 bbuf[16];
        uint4 u0 = *(const uint4*)(kb);
        uint4 u1 = *(const uint4*)(kb + 16);
        __builtin_memcpy(bbuf,     &u0, 16);
        __builtin_memcpy(bbuf + 8, &u1, 16);
        v16bf Bk; __builtin_memcpy(&Bk, bbuf, 32);
        v8f c = {};
        c = __builtin_amdgcn_wmma_f32_16x16x32_bf16(false, Aq, false, Bk,
                                                    (short)0, c, false, false);
#pragma unroll
        for (int i = 0; i < 8; ++i) {
            int nl = i + 8 * lh;
            probs[nl * LDSTR + mt * 16 + ll] =
                c[i] + biasrow[(size_t)(n0 + nl) * Nn + (mt * 16 + ll)];
        }
    }
    __syncthreads();

    // ---- stage 2: row softmax; normalization AND edge-mask folded into probs -
    {
        int row = tid & 15, s = tid >> 4;          // s in 0..5
        float mx = -3.4e38f;
        for (int m = s; m < Nn; m += 6) mx = fmaxf(mx, probs[row * LDSTR + m]);
        red[row * 6 + s] = mx;
        __syncthreads();
        if (s == 0) {
            float m2 = red[row * 6];
            for (int j = 1; j < 6; ++j) m2 = fmaxf(m2, red[row * 6 + j]);
            rowstat[row] = m2;
        }
        __syncthreads();
        float rm = rowstat[row], sum = 0.f;
        for (int m = s; m < Nn; m += 6) {
            float p = __expf(probs[row * LDSTR + m] - rm);
            probs[row * LDSTR + m] = p;
            sum += p;
        }
        __syncthreads();
        red[row * 6 + s] = sum;
        __syncthreads();
        if (s == 0) {
            float t = 0.f;
            for (int j = 0; j < 6; ++j) t += red[row * 6 + j];
            rowstat[row] = 1.0f / t;
        }
        wait_async<0>();                           // mask tile (and chunk 0) landed
        __syncthreads();
        // scale pass: probs = softmax prob, zeroed where edge is dropped
        float inv = rowstat[row];
        const unsigned char* mr = mbuf + row * MROW;
        for (int m = s; m < Nn; m += 6) {
            float p = probs[row * LDSTR + m] * inv;
            if (doa && mr[m]) p = 0.f;
            probs[row * LDSTR + m] = p;
        }
    }
    __syncthreads();

    // ---- stage 3: (probs * delta[...,c]) @ v, K = 512 ------------------------
    const int c_idx = wave;
    v8f acc0 = {}, acc1 = {};                       // d = 0..15 / 16..31
    const __bf16* vbase = v_t + (size_t)bh * Dd * Nn;
    const int nl0 = tid / 24, o16 = tid % 24;
    for (int mt = 0; mt < 32; ++mt) {
        if (mt < 31) {                              // prefetch next delta chunk
            float* nb = dbuf[(mt + 1) & 1];
#pragma unroll
            for (int r = 0; r < 4; ++r) {
                int nl = nl0 + 4 * r;
                cp16(dpb + (size_t)nl * (Nn * 3) + (mt + 1) * 96 + o16 * 4,
                     &nb[nl * DROW + o16 * 4]);
            }
            wait_async<4>();                        // chunk mt complete (in-order)
        } else {
            wait_async<0>();
        }
        __syncthreads();

        const float* db = dbuf[mt & 1] + ll * DROW;
        const float* pr = &probs[ll * LDSTR + mt * 32];
        __bf16 abuf[16];
#pragma unroll
        for (int e = 0; e < 8; e += 2) {
            int ml = lh * 8 + e;                    // local m in chunk
            float2 p2 = *(const float2*)(pr + ml);
            pack2(p2.x * db[ml * 3 + c_idx], p2.y * db[ml * 3 + 3 + c_idx], abuf + e);
        }
#pragma unroll
        for (int e = 0; e < 8; e += 2) {
            int ml = lh * 8 + 16 + e;
            float2 p2 = *(const float2*)(pr + ml);
            pack2(p2.x * db[ml * 3 + c_idx], p2.y * db[ml * 3 + 3 + c_idx], abuf + 8 + e);
        }
        v16bf Ar; __builtin_memcpy(&Ar, abuf, 32);

        const int mcol = mt * 32 + lh * 8;
        __bf16 bbuf[16];
        v16bf B0, B1;
        const __bf16* p0 = vbase + (size_t)ll * Nn + mcol;          // d = ll
        uint4 u0 = *(const uint4*)(p0);
        uint4 u1 = *(const uint4*)(p0 + 16);
        __builtin_memcpy(bbuf, &u0, 16); __builtin_memcpy(bbuf + 8, &u1, 16);
        __builtin_memcpy(&B0, bbuf, 32);
        const __bf16* p1 = vbase + (size_t)(16 + ll) * Nn + mcol;   // d = 16+ll
        uint4 u2 = *(const uint4*)(p1);
        uint4 u3 = *(const uint4*)(p1 + 16);
        __builtin_memcpy(bbuf, &u2, 16); __builtin_memcpy(bbuf + 8, &u3, 16);
        __builtin_memcpy(&B1, bbuf, 32);

        acc0 = __builtin_amdgcn_wmma_f32_16x16x32_bf16(false, Ar, false, B0,
                                                       (short)0, acc0, false, false);
        acc1 = __builtin_amdgcn_wmma_f32_16x16x32_bf16(false, Ar, false, B1,
                                                       (short)0, acc1, false, false);
        __syncthreads();
    }

    // ---- stage 4: rank-1 projection partial dot, atomic accumulate over h ----
    const float* Wf = (c_idx == 0) ? Wf1 : (c_idx == 1) ? Wf2 : Wf3;
    const float w0 = Wf[h * Dd + ll];
    const float w1 = Wf[h * Dd + 16 + ll];
#pragma unroll
    for (int i = 0; i < 8; ++i) {
        float s = acc0[i] * w0 + acc1[i] * w1;
        s += __shfl_xor(s, 1, 32);
        s += __shfl_xor(s, 2, 32);
        s += __shfl_xor(s, 4, 32);
        s += __shfl_xor(s, 8, 32);
        if (ll == 0) {
            int nl = i + 8 * lh;
            atomicAdd(&out[((size_t)(b * Nn) + n0 + nl) * 3 + c_idx], s);
        }
    }
}

// ---------------------------------------------------------------------------
extern "C" void kernel_launch(void* const* d_in, const int* in_sizes, int n_in,
                              void* d_out, int out_size, void* d_ws, size_t ws_size,
                              hipStream_t stream) {
    const float* query     = (const float*)d_in[0];
    const float* attn_bias = (const float*)d_in[1];
    const float* delta_pos = (const float*)d_in[2];
    const unsigned char* dmask = (const unsigned char*)d_in[3];
    const float* Wq = (const float*)d_in[4];
    const float* bq = (const float*)d_in[5];
    const float* Wk = (const float*)d_in[6];
    const float* bk = (const float*)d_in[7];
    const float* Wv = (const float*)d_in[8];
    const float* bv = (const float*)d_in[9];
    const float* Wf1 = (const float*)d_in[10];
    const float* bf1 = (const float*)d_in[11];
    const float* Wf2 = (const float*)d_in[12];
    const float* bf2 = (const float*)d_in[13];
    const float* Wf3 = (const float*)d_in[14];
    const float* bf3 = (const float*)d_in[15];
    const int*   doa = (const int*)d_in[16];

    char* ws = (char*)d_ws;                    // 3 MB of bf16 q / k / v^T
    __bf16* q_s = (__bf16*)(ws);
    __bf16* k_m = (__bf16*)(ws + (1 << 20));
    __bf16* v_t = (__bf16*)(ws + (2 << 20));
    float* out = (float*)d_out;

    qkv_kernel<<<dim3(128, 16, 3), dim3(32), 0, stream>>>(
        query, Wq, bq, Wk, bk, Wv, bv, q_s, k_m, v_t);
    init_out_kernel<<<dim3((Bb * Nn * 3 + 255) / 256), dim3(256), 0, stream>>>(
        out, bf1, bf2, bf3);
    attn_fused_kernel<<<dim3(Nn / 16, Bb * Hh), dim3(96), 0, stream>>>(
        q_s, k_m, v_t, attn_bias, delta_pos, dmask, doa, Wf1, Wf2, Wf3, out);
}
